// GATRatingPrediction_44074954392215
// MI455X (gfx1250) — compile-verified, matrix operand order checked by reading, not run on hardware
//
#include <hip/hip_runtime.h>
#include <hip/hip_bf16.h>
#include <stdint.h>

typedef __bf16 bf16_t;
typedef __attribute__((ext_vector_type(16))) __bf16 v16bf;
typedef __attribute__((ext_vector_type(8)))  float  v8f;

#define NEG_SLOPE 0.2f

// order-preserving monotonic encoding of f32 into u32 (for atomicMax)
__device__ __forceinline__ unsigned enc_f32(float f) {
    unsigned u = __float_as_uint(f);
    return (u & 0x80000000u) ? ~u : (u | 0x80000000u);
}
__device__ __forceinline__ float dec_f32(unsigned u) {
    return (u & 0x80000000u) ? __uint_as_float(u & 0x7fffffffu)
                             : __uint_as_float(~u);
}
#define ENC_NEG_INF 0x007FFFFFu  // enc(-inf)

// ---------------- fill ----------------
__global__ void fill_u32(unsigned* __restrict__ p, unsigned v, long long n) {
    long long i = (long long)blockIdx.x * blockDim.x + threadIdx.x;
    if (i < n) p[i] = v;
}

// ---- pre-swizzle weight matrix W[K][Nout] (row-major f32) into the exact
// ---- B-fragment layout of V_WMMA_F32_16X16X32_BF16 (ISA 7.12.2):
// ---- frag[(nt*(K/32)+ks)*32 + lane][j] = W[ks*32 + (lane>>4)*8 + (j<8?j:8+j)][nt*16 + (lane&15)]
__global__ void prep_wfrag(const float* __restrict__ W, bf16_t* __restrict__ F,
                           int K, int Nout) {
    int idx = blockIdx.x * blockDim.x + threadIdx.x;
    if (idx >= K * Nout) return;
    int j    = idx & 15;
    int lane = (idx >> 4) & 31;
    int rest = idx >> 9;
    int kt   = K >> 5;
    int ks   = rest % kt;
    int nt   = rest / kt;
    int n    = (nt << 4) + (lane & 15);
    int kgrp = (lane >> 4) << 3;
    int k    = (ks << 5) + kgrp + (j < 8 ? j : 8 + j);
    F[idx] = (bf16_t)W[(size_t)k * Nout + n];
}

// ---- one wave computes one 16x16 f32 tile of C = A[M,K] * W[K,Nout] ----
// requires M % 16 == 0 (M = 50000 = 3125*16). EXEC stays all-ones (WMMA req).
__global__ __launch_bounds__(32)
void gemm_wmma_bf16(const float* __restrict__ A, const bf16_t* __restrict__ Bfrag,
                    float* __restrict__ C, int K, int Nout) {
    const int lane = threadIdx.x;
    const int mt = blockIdx.x, nt = blockIdx.y;
    const int m    = (mt << 4) + (lane & 15);
    const int kgrp = (lane >> 4) << 3;
    const int kt   = K >> 5;
    v8f acc = {};
    const float* arow = A + (size_t)m * K + kgrp;
    const v16bf* bptr = (const v16bf*)Bfrag + (size_t)nt * kt * 32 + lane;
    for (int ks = 0; ks < kt; ++ks) {
        __builtin_prefetch(arow + 32, 0, 3);   // global_prefetch for next K-step
        v16bf a;
        v16bf b = bptr[(size_t)ks * 32];
        #pragma unroll
        for (int j = 0; j < 8; ++j)  a[j] = (bf16_t)arow[j];        // K = kgrp+0..7
        #pragma unroll
        for (int j = 8; j < 16; ++j) a[j] = (bf16_t)arow[8 + j];    // K = kgrp+16..23
        acc = __builtin_amdgcn_wmma_f32_16x16x32_bf16(
                  false, a, false, b, (short)0, acc, false, false);
        arow += 32;
    }
    const int col   = (nt << 4) + (lane & 15);
    const int rbase = (mt << 4) + ((lane >> 4) << 3);   // lanes 16-31 hold M=8..15
    #pragma unroll
    for (int r = 0; r < 8; ++r)
        C[(size_t)(rbase + r) * Nout + col] = acc[r];
}

// ---- per (node,head) attention logits: es = <h, a_src>, ed = <h, a_dst> ----
__global__ void attn_scores(const float* __restrict__ h, const float* __restrict__ asrc,
                            const float* __restrict__ adst, float* __restrict__ es,
                            float* __restrict__ ed, int NH, int H) {
    int i = blockIdx.x * blockDim.x + threadIdx.x;
    if (i >= NH) return;
    int hh = i % H;
    const float* hp = h + (size_t)i * 32;
    const float* ap = asrc + hh * 32;
    const float* dp = adst + hh * 32;
    float s = 0.f, d = 0.f;
    #pragma unroll
    for (int c = 0; c < 32; ++c) { float v = hp[c]; s += v * ap[c]; d += v * dp[c]; }
    es[i] = s; ed[i] = d;
}

__device__ __forceinline__ void edge_ends(int e, int E, const int* src, const int* dst,
                                          int& s, int& d) {
    if (e < E) { s = src[e]; d = dst[e]; } else { s = d = e - E; }  // self loops
}

// ---- pass 1: segment max of leaky(e_src[src]+e_dst[dst]) into menc[dst,h] ----
__global__ void edge_max_k(const int* __restrict__ src, const int* __restrict__ dst,
                           const float* __restrict__ es, const float* __restrict__ ed,
                           unsigned* __restrict__ menc, int E, int N, int H) {
    int e = blockIdx.x * blockDim.x + threadIdx.x;
    if (e >= E + N) return;
    int s, d; edge_ends(e, E, src, dst, s, d);
    for (int h = 0; h < H; ++h) {
        float v = es[s * H + h] + ed[d * H + h];
        v = (v > 0.f) ? v : v * NEG_SLOPE;
        atomicMax(&menc[d * H + h], enc_f32(v));
    }
}

// ---- pass 2: segment sum of exp(e - m) into ssum[dst,h] ----
__global__ void edge_sum_k(const int* __restrict__ src, const int* __restrict__ dst,
                           const float* __restrict__ es, const float* __restrict__ ed,
                           const unsigned* __restrict__ menc, float* __restrict__ ssum,
                           int E, int N, int H) {
    int e = blockIdx.x * blockDim.x + threadIdx.x;
    if (e >= E + N) return;
    int s, d; edge_ends(e, E, src, dst, s, d);
    for (int h = 0; h < H; ++h) {
        float v = es[s * H + h] + ed[d * H + h];
        v = (v > 0.f) ? v : v * NEG_SLOPE;
        atomicAdd(&ssum[d * H + h], __expf(v - dec_f32(menc[d * H + h])));
    }
}

// ---- pass 3: out[dst] += alpha * h[src] ; blockDim.x = 256, 256/HC edges per block ----
__global__ void edge_agg_k(const int* __restrict__ src, const int* __restrict__ dst,
                           const float* __restrict__ es, const float* __restrict__ ed,
                           const unsigned* __restrict__ menc, const float* __restrict__ ssum,
                           const float* __restrict__ hin, float* __restrict__ out,
                           int E, int N, int H, int HC) {
    int le   = threadIdx.x / HC;
    int elem = threadIdx.x - le * HC;
    int e    = blockIdx.x * (blockDim.x / HC) + le;
    if (e >= E + N) return;
    int s, d; edge_ends(e, E, src, dst, s, d);
    int h = elem >> 5;
    float v = es[s * H + h] + ed[d * H + h];
    v = (v > 0.f) ? v : v * NEG_SLOPE;
    float alpha = __expf(v - dec_f32(menc[d * H + h])) / (ssum[d * H + h] + 1e-16f);
    atomicAdd(&out[(size_t)d * HC + elem], hin[(size_t)s * HC + elem] * alpha);
}

// ---- out = relu(out + bias) in place ----
__global__ void bias_relu_k(float* __restrict__ buf, const float* __restrict__ b,
                            long long total, int HC) {
    long long i = (long long)blockIdx.x * blockDim.x + threadIdx.x;
    if (i >= total) return;
    float v = buf[i] + b[i % HC];
    buf[i] = v > 0.f ? v : 0.f;
}

// ---- predictor: sigmoid([h3[src]|h3[dst]] . Wp + bp) * 4 + 1 ----
__global__ void predict_k(const int* __restrict__ src, const int* __restrict__ dst,
                          const float* __restrict__ h3, const float* __restrict__ Wp,
                          const float* __restrict__ bp, float* __restrict__ out, int E) {
    int e = blockIdx.x * blockDim.x + threadIdx.x;
    if (e >= E) return;
    const float* hs = h3 + (size_t)src[e] * 32;
    const float* hd = h3 + (size_t)dst[e] * 32;
    float acc = bp[0];
    #pragma unroll
    for (int c = 0; c < 32; ++c) acc += hs[c] * Wp[c];
    #pragma unroll
    for (int c = 0; c < 32; ++c) acc += hd[c] * Wp[32 + c];
    out[e] = 4.0f / (1.0f + __expf(-acc)) + 1.0f;
}

// =================== host-side orchestration ===================
static void run_layer(const float* Ain, const bf16_t* wf, int K, int H, int HC,
                      const float* as, const float* ad, const float* b,
                      const int* srcp, const int* dstp, int E, int N,
                      float* hG, float* hA, float* es, float* ed,
                      unsigned* menc, float* ssum, hipStream_t stream) {
    gemm_wmma_bf16<<<dim3(N / 16, HC / 16), 32, 0, stream>>>(Ain, wf, hG, K, HC);
    int NH = N * H;
    attn_scores<<<(NH + 255) / 256, 256, 0, stream>>>(hG, as, ad, es, ed, NH, H);
    fill_u32<<<(NH + 255) / 256, 256, 0, stream>>>(menc, ENC_NEG_INF, NH);
    fill_u32<<<(NH + 255) / 256, 256, 0, stream>>>((unsigned*)ssum, 0u, NH);
    long long tot = (long long)N * HC;
    fill_u32<<<(unsigned)((tot + 255) / 256), 256, 0, stream>>>((unsigned*)hA, 0u, tot);
    int ET = E + N;
    edge_max_k<<<(ET + 255) / 256, 256, 0, stream>>>(srcp, dstp, es, ed, menc, E, N, H);
    edge_sum_k<<<(ET + 255) / 256, 256, 0, stream>>>(srcp, dstp, es, ed, menc, ssum, E, N, H);
    int epb = 256 / HC;
    edge_agg_k<<<(ET + epb - 1) / epb, 256, 0, stream>>>(srcp, dstp, es, ed, menc, ssum,
                                                         hG, hA, E, N, H, HC);
    bias_relu_k<<<(unsigned)((tot + 255) / 256), 256, 0, stream>>>(hA, b, tot, HC);
}

extern "C" void kernel_launch(void* const* d_in, const int* in_sizes, int n_in,
                              void* d_out, int out_size, void* d_ws, size_t ws_size,
                              hipStream_t stream) {
    const float* x   = (const float*)d_in[0];
    const int*   ei  = (const int*)d_in[1];
    const float* W1  = (const float*)d_in[2];
    const float* as1 = (const float*)d_in[3];
    const float* ad1 = (const float*)d_in[4];
    const float* b1  = (const float*)d_in[5];
    const float* W2  = (const float*)d_in[6];
    const float* as2 = (const float*)d_in[7];
    const float* ad2 = (const float*)d_in[8];
    const float* b2  = (const float*)d_in[9];
    const float* W3  = (const float*)d_in[10];
    const float* as3 = (const float*)d_in[11];
    const float* ad3 = (const float*)d_in[12];
    const float* b3  = (const float*)d_in[13];
    const float* Wp  = (const float*)d_in[14];
    const float* bp  = (const float*)d_in[15];
    float* out = (float*)d_out;
    (void)n_in; (void)out_size; (void)ws_size;

    const int N = in_sizes[0] / 128;   // 50000 (multiple of 16)
    const int E = in_sizes[1] / 2;     // 800000
    const int* srcp = ei;
    const int* dstp = ei + E;

    char* ws = (char*)d_ws;
    size_t off = 0;
    auto alloc = [&](size_t bytes) -> char* {
        char* p = ws + off; off += (bytes + 255) & ~(size_t)255; return p;
    };
    float*    hG   = (float*)alloc((size_t)N * 256 * 4);   // GEMM outputs h (pre-bias)
    float*    hA   = (float*)alloc((size_t)N * 256 * 4);   // aggregated / layer activations
    float*    es   = (float*)alloc((size_t)N * 8 * 4);
    float*    ed   = (float*)alloc((size_t)N * 8 * 4);
    unsigned* menc = (unsigned*)alloc((size_t)N * 8 * 4);
    float*    ssum = (float*)alloc((size_t)N * 8 * 4);
    bf16_t*   wf1  = (bf16_t*)alloc((size_t)128 * 256 * 2);
    bf16_t*   wf2  = (bf16_t*)alloc((size_t)256 * 128 * 2);
    bf16_t*   wf3  = (bf16_t*)alloc((size_t)128 * 32 * 2);

    // pre-swizzle weights into WMMA B-fragment layout (bf16)
    prep_wfrag<<<(128 * 256 + 255) / 256, 256, 0, stream>>>(W1, wf1, 128, 256);
    prep_wfrag<<<(256 * 128 + 255) / 256, 256, 0, stream>>>(W2, wf2, 256, 128);
    prep_wfrag<<<(128 * 32 + 255) / 256, 256, 0, stream>>>(W3, wf3, 128, 32);

    // conv1: in=128, heads=8, HC=256
    run_layer(x,  wf1, 128, 8, 256, as1, ad1, b1, srcp, dstp, E, N,
              hG, hA, es, ed, menc, ssum, stream);
    // conv2: in=256, heads=4, HC=128  (input = hA, GEMM output hG — no alias hazard)
    run_layer(hA, wf2, 256, 4, 128, as2, ad2, b2, srcp, dstp, E, N,
              hG, hA, es, ed, menc, ssum, stream);
    // conv3: in=128, heads=1, HC=32 (mean over 1 head == identity)
    run_layer(hA, wf3, 128, 1, 32, as3, ad3, b3, srcp, dstp, E, N,
              hG, hA, es, ed, menc, ssum, stream);

    predict_k<<<(E + 255) / 256, 256, 0, stream>>>(srcp, dstp, hA, Wp, bp, out, E);
}
